// EncoderLayer_hybrid2_8315056685232
// MI455X (gfx1250) — matrix operand order repaired and checked by reading, not compiled, not run on hardware
//
#include <hip/hip_runtime.h>

// EncoderLayer hybrid attention for MI455X (gfx1250), wave32 + WMMA +
// async global->LDS double-buffering. B=8, T=1024, E=512, H=8, EH=64.

#define B_  8
#define T_  1024
#define E_  512
#define H_  8
#define EH_ 64

typedef __attribute__((ext_vector_type(16))) _Float16 v16h;
typedef __attribute__((ext_vector_type(8)))  _Float16 v8h;
typedef __attribute__((ext_vector_type(8)))  float    v8f;

static __device__ __forceinline__ v16h cat16(v8h lo, v8h hi)
{
    return __builtin_shufflevector(lo, hi, 0,1,2,3,4,5,6,7,8,9,10,11,12,13,14,15);
}

// CDNA5 async DMA: memory -> LDS, tracked by ASYNCcnt (ISA 15.18.3 / 5.7.1).
// dst_lds = LDS byte offset (low 32 bits of flat shared address per ISA 10.2),
// src     = 64-bit global address (prints as a VGPR pair).
static __device__ __forceinline__ void async_copy_b128(unsigned dst_lds, const void* src)
{
    asm volatile("global_load_async_to_lds_b128 %0, %1, off"
                 :: "v"(dst_lds), "v"(src) : "memory");
}
static __device__ __forceinline__ void wait_async0()
{
    asm volatile("s_wait_asynccnt 0x0" ::: "memory");
}

// ---------------------------------------------------------------------------
// Kernel 1: LayerNorm + scalar Q/K per (token, head); store xn as f16 in
// layout [b,h,t,eh] so each (b,h) slice is a contiguous 1024x64 A-matrix.
// ---------------------------------------------------------------------------
__global__ __launch_bounds__(128) void ln_qk_kernel(
    const float* __restrict__ x,
    const float* __restrict__ W_Q, const float* __restrict__ b_Q,
    const float* __restrict__ W_K, const float* __restrict__ b_K,
    _Float16* __restrict__ xn_ws, float* __restrict__ Qb, float* __restrict__ Kb)
{
    __shared__ float red[128];
    __shared__ float xn_s[E_];
    const int token = blockIdx.x;           // b*T + t
    const int b = token / T_, t = token % T_;
    const int tid = threadIdx.x;

    const float* xp = x + (size_t)token * E_;
    float4 v = ((const float4*)xp)[tid];    // channels tid*4 .. tid*4+3
    float s  = v.x + v.y + v.z + v.w;
    float ss = v.x*v.x + v.y*v.y + v.z*v.z + v.w*v.w;

    red[tid] = s; __syncthreads();
    for (int st = 64; st > 0; st >>= 1) { if (tid < st) red[tid] += red[tid + st]; __syncthreads(); }
    float mu = red[0] * (1.0f / E_);
    __syncthreads();
    red[tid] = ss; __syncthreads();
    for (int st = 64; st > 0; st >>= 1) { if (tid < st) red[tid] += red[tid + st]; __syncthreads(); }
    float var  = red[0] * (1.0f / E_) - mu * mu;
    float rstd = rsqrtf(var + 1e-5f);

    int e0 = tid * 4;
    int h  = e0 / EH_, eh = e0 % EH_;
    float xn0 = (v.x - mu) * rstd, xn1 = (v.y - mu) * rstd;
    float xn2 = (v.z - mu) * rstd, xn3 = (v.w - mu) * rstd;
    xn_s[e0+0] = xn0; xn_s[e0+1] = xn1; xn_s[e0+2] = xn2; xn_s[e0+3] = xn3;
    _Float16* xp16 = xn_ws + (((size_t)(b*H_ + h) * T_ + t) * EH_ + eh);
    xp16[0] = (_Float16)xn0; xp16[1] = (_Float16)xn1;
    xp16[2] = (_Float16)xn2; xp16[3] = (_Float16)xn3;
    __syncthreads();

    // scalar Q/K dots: 16 threads per head, 4 channels each, shuffle-reduce
    int h2 = tid >> 4, ln = tid & 15;
    float pq = 0.f, pk = 0.f;
    #pragma unroll
    for (int c = 0; c < 4; ++c) {
        int ee = h2 * EH_ + ln * 4 + c;
        float xv = xn_s[ee];
        pq = fmaf(xv, W_Q[ee], pq);
        pk = fmaf(xv, W_K[ee], pk);
    }
    #pragma unroll
    for (int d = 8; d > 0; d >>= 1) {
        pq += __shfl_down(pq, d, 16);
        pk += __shfl_down(pk, d, 16);
    }
    if (ln == 0) {
        Qb[(size_t)(b*H_ + h2) * T_ + t] = pq + b_Q[h2];
        Kb[(size_t)(b*H_ + h2) * T_ + t] = pk + b_K[h2];
    }
}

// ---------------------------------------------------------------------------
// Kernel 2: per-head V projection via WMMA. One wave = one 16x64 tile.
// W_V[h] staged TRANSPOSED ([n][k]) in LDS so each B-fragment is two
// contiguous ds_load_b128 per lane; all four fragments fetched before the
// WMMA quartet so LDS loads stream while the matrix pipe drains.
// V stored f-major [b,h,f,t] for the attention GEMM.
// ---------------------------------------------------------------------------
__global__ __launch_bounds__(256) void v_gemm_kernel(
    const _Float16* __restrict__ xn_ws, const float* __restrict__ W_V,
    const float* __restrict__ b_V, _Float16* __restrict__ Vt_ws)
{
    __shared__ __align__(16) _Float16 WvT[EH_ * EH_];  // [n][k] f16, 8 KB
    const int bh  = blockIdx.x >> 3;                   // 8 blocks per (b,h)
    const int blk = blockIdx.x & 7;
    const int h   = bh & (H_ - 1);
    const int tid = threadIdx.x;

    // stage W_V[h] (row-major [k][n] f32) transposed into LDS as [n][k] f16
    const float* wv = W_V + (size_t)h * EH_ * EH_;
    for (int i = tid; i < EH_ * EH_; i += 256) {
        int k = i >> 6, n = i & 63;
        WvT[n * EH_ + k] = (_Float16)wv[i];
    }
    __syncthreads();

    const int w    = tid >> 5, lane = tid & 31;
    const int t0   = blk * 128 + w * 16;
    const int mrow = lane & 15;
    const int hi   = lane >> 4;                        // lane-half selects K split
    const _Float16* abase = xn_ws + ((size_t)bh * T_ + (t0 + mrow)) * EH_;

    v8f acc[4] = {};
    #pragma unroll
    for (int s = 0; s < 2; ++s) {                      // K = 64 in 2 steps of 32
        v16h a = cat16(*(const v8h*)(abase + s*32 + hi*8),
                       *(const v8h*)(abase + s*32 + hi*8 + 16));
        const int kb = s*32 + hi*8;
        v16h bf[4];
        #pragma unroll
        for (int nt = 0; nt < 4; ++nt) {               // fetch all B-fragments
            const _Float16* bp = WvT + (nt*16 + (lane & 15)) * EH_ + kb;
            bf[nt] = cat16(*(const v8h*)bp, *(const v8h*)(bp + 16));
        }
        #pragma unroll
        for (int nt = 0; nt < 4; ++nt)                 // then the WMMA quartet
            acc[nt] = __builtin_amdgcn_wmma_f32_16x16x32_f16(
                false, a, false, bf[nt], (short)0, acc[nt], false, false);
    }
    #pragma unroll
    for (int nt = 0; nt < 4; ++nt) {
        float bv = b_V[h * EH_ + nt*16 + (lane & 15)];
        #pragma unroll
        for (int r = 0; r < 8; ++r) {
            int row = t0 + r + hi*8;                   // token j
            int col = nt*16 + (lane & 15);             // feature f
            Vt_ws[((size_t)bh * EH_ + col) * T_ + row] = (_Float16)(acc[nt][r] + bv);
        }
    }
}

// ---------------------------------------------------------------------------
// Kernel 3: softmax row statistics. Scores are scalar: s_ij = -(Q_i-K_j)^2/8.
// K vector (4 KB) lives in LDS; each thread scans 1024 entries twice.
// ---------------------------------------------------------------------------
__global__ __launch_bounds__(256) void stats_kernel(
    const float* __restrict__ Qb, const float* __restrict__ Kb,
    float* __restrict__ Mb, float* __restrict__ Lb)
{
    __shared__ float ks[T_];
    const int bh    = blockIdx.x >> 2;
    const int chunk = blockIdx.x & 3;
    const int tid   = threadIdx.x;
    const float* kp = Kb + (size_t)bh * T_;
    for (int j = tid; j < T_; j += 256) ks[j] = kp[j];
    __syncthreads();

    const int i = chunk * 256 + tid;
    float q = Qb[(size_t)bh * T_ + i];
    float m = -3.0e38f;
    for (int j = 0; j < T_; ++j) {
        float d = q - ks[j];
        m = fmaxf(m, -d * d * 0.125f);
    }
    float l = 0.f;
    for (int j = 0; j < T_; ++j) {
        float d = q - ks[j];
        l += __expf(fmaf(-d * 0.125f, d, -m));
    }
    Mb[(size_t)bh * T_ + i] = m;
    Lb[(size_t)bh * T_ + i] = l;
}

// ---------------------------------------------------------------------------
// Kernel 4: flash-style attn @ V with WMMA. Wave owns 16x64 output tile;
// probability A-fragments recomputed in registers (TRANS exp co-executes
// with WMMA per ISA 8.4). V chunks (64 keys x 64 feats, f-major so K is
// contiguous) are DMA'd memory->LDS with GLOBAL_LOAD_ASYNC_TO_LDS_B128 and
// double-buffered: issue chunk n+1, compute chunk n while the async engine
// fills the other buffer (ASYNCcnt / s_wait_asynccnt pipeline).
// ---------------------------------------------------------------------------
__global__ __launch_bounds__(256) void attn_v_kernel(
    const float* __restrict__ Qb, const float* __restrict__ Kb,
    const float* __restrict__ Mb, const float* __restrict__ Lb,
    const _Float16* __restrict__ Vt_ws, float* __restrict__ HO)
{
    __shared__ float ks[T_];                              // 4 KB
    __shared__ __align__(16) _Float16 vch[2][EH_ * 64];   // 2 x 8 KB, [f][j]
    const int bh  = blockIdx.x >> 3;
    const int blk = blockIdx.x & 7;
    const int tid = threadIdx.x;
    const float* kp = Kb + (size_t)bh * T_;
    for (int j = tid; j < T_; j += 256) ks[j] = kp[j];

    const int w    = tid >> 5, lane = tid & 31;
    const int hi   = lane >> 4;
    const int i0   = blk * 128 + w * 16;
    const int irow = i0 + (lane & 15);
    float q    = Qb[(size_t)bh * T_ + irow];
    float mi   = Mb[(size_t)bh * T_ + irow];
    float invl = 1.0f / Lb[(size_t)bh * T_ + irow];

    v8f acc[4] = {};
    const _Float16* vbase = Vt_ws + (size_t)bh * EH_ * T_;   // [f][t]

    // staging coordinates for this thread: feature row + 16-key segment
    const int sf = tid >> 2;            // 0..63
    const int sj = (tid & 3) * 16;      // 0,16,32,48

    auto issue_chunk = [&](int jb, int pbuf) {
        const _Float16* src = vbase + (size_t)sf * T_ + jb + sj;
        unsigned dst = (unsigned)(size_t)(&vch[pbuf][sf * 64 + sj]);
        async_copy_b128(dst,      src);
        async_copy_b128(dst + 16, src + 8);
    };

    issue_chunk(0, 0);                  // prologue: DMA chunk 0

    for (int jb = 0; jb < T_; jb += 64) {
        const int p = (jb >> 6) & 1;
        wait_async0();                  // my DMA for chunk jb has landed
        __syncthreads();                // everyone's landed; buffer p^1 free
        if (jb + 64 < T_) issue_chunk(jb + 64, p ^ 1);   // overlap next DMA

        const _Float16* vcur = vch[p];
        #pragma unroll
        for (int s = 0; s < 2; ++s) {   // two K=32 steps within the 64-chunk
            // probability A-fragment: p = exp(score - m) / l, f16
            v16h a;
            int jbase = jb + s*32 + hi*8;
            #pragma unroll
            for (int k = 0; k < 8; ++k) {
                float d0 = q - ks[jbase + k];
                a[k]   = (_Float16)(__expf(fmaf(-d0 * 0.125f, d0, -mi)) * invl);
                float d1 = q - ks[jbase + 16 + k];
                a[8+k] = (_Float16)(__expf(fmaf(-d1 * 0.125f, d1, -mi)) * invl);
            }
            const int kb = s*32 + hi*8;
            v16h bf[4];
            #pragma unroll
            for (int nt = 0; nt < 4; ++nt) {           // fetch all B-fragments
                const _Float16* bp = vcur + (nt*16 + (lane & 15)) * 64 + kb;
                bf[nt] = cat16(*(const v8h*)bp, *(const v8h*)(bp + 16));
            }
            #pragma unroll
            for (int nt = 0; nt < 4; ++nt)             // then the WMMA quartet
                acc[nt] = __builtin_amdgcn_wmma_f32_16x16x32_f16(
                    false, a, false, bf[nt], (short)0, acc[nt], false, false);
        }
    }
    #pragma unroll
    for (int nt = 0; nt < 4; ++nt) {
        #pragma unroll
        for (int r = 0; r < 8; ++r) {
            int row = i0 + r + hi*8;
            int col = nt*16 + (lane & 15);
            HO[((size_t)bh * T_ + row) * EH_ + col] = acc[nt][r];
        }
    }
}

// ---------------------------------------------------------------------------
// Kernel 5: 8x8 merger MLP over head axis + residual. out[e = f*H + h].
// ---------------------------------------------------------------------------
__global__ __launch_bounds__(256) void merger_kernel(
    const float* __restrict__ HO, const float* __restrict__ x,
    const float* __restrict__ W1, const float* __restrict__ b1,
    const float* __restrict__ W2, const float* __restrict__ b2,
    float* __restrict__ out)
{
    __shared__ float w1s[H_*H_], w2s[H_*H_], b1s[H_], b2s[H_];
    int tid = threadIdx.x;
    if (tid < 64)       w1s[tid]       = W1[tid];
    else if (tid < 128) w2s[tid - 64]  = W2[tid - 64];
    else if (tid < 136) b1s[tid - 128] = b1[tid - 128];
    else if (tid < 144) b2s[tid - 136] = b2[tid - 136];
    __syncthreads();

    size_t g = (size_t)blockIdx.x * 256 + tid;   // over B*T*EH
    int f = (int)(g & 63);
    int i = (int)((g >> 6) & (T_ - 1));
    int b = (int)(g >> 16);                      // 64*1024 per batch

    float v[H_];
    #pragma unroll
    for (int h = 0; h < H_; ++h)
        v[h] = HO[(((size_t)(b*H_ + h)) * T_ + i) * EH_ + f];

    float h1[H_];
    #pragma unroll
    for (int gg = 0; gg < H_; ++gg) {
        float s = b1s[gg];
        #pragma unroll
        for (int h = 0; h < H_; ++h) s = fmaf(v[h], w1s[h*H_ + gg], s);
        h1[gg] = fmaxf(s, 0.f);
    }
    const float* xp = x   + ((size_t)(b*T_ + i)) * E_ + f * H_;
    float*       op = out + ((size_t)(b*T_ + i)) * E_ + f * H_;
    #pragma unroll
    for (int kk = 0; kk < H_; ++kk) {
        float s = b2s[kk];
        #pragma unroll
        for (int gg = 0; gg < H_; ++gg) s = fmaf(h1[gg], w2s[gg*H_ + kk], s);
        op[kk] = s + xp[kk];
    }
}

// ---------------------------------------------------------------------------
extern "C" void kernel_launch(void* const* d_in, const int* in_sizes, int n_in,
                              void* d_out, int out_size, void* d_ws, size_t ws_size,
                              hipStream_t stream)
{
    (void)in_sizes; (void)n_in; (void)out_size; (void)ws_size;
    const float* x   = (const float*)d_in[0];
    const float* W_V = (const float*)d_in[1];
    const float* b_V = (const float*)d_in[2];
    const float* W_Q = (const float*)d_in[3];
    const float* b_Q = (const float*)d_in[4];
    const float* W_K = (const float*)d_in[5];
    const float* b_K = (const float*)d_in[6];
    const float* W1  = (const float*)d_in[7];
    const float* b1  = (const float*)d_in[8];
    const float* W2  = (const float*)d_in[9];
    const float* b2  = (const float*)d_in[10];
    float* out = (float*)d_out;

    char* ws = (char*)d_ws;
    _Float16* xn = (_Float16*)(ws);                    // 8 MB  [b,h,t,eh] f16
    _Float16* Vt = (_Float16*)(ws + 8388608);          // 8 MB  [b,h,f,t]  f16
    float* Qb = (float*)(ws + 16777216);               // 256 KB
    float* Kb = (float*)(ws + 17039360);               // 256 KB
    float* Mb = (float*)(ws + 17301504);               // 256 KB
    float* Lb = (float*)(ws + 17563648);               // 256 KB
    float* HO = (float*)(ws + 17825792);               // 16 MB [b,h,t,f] f32

    ln_qk_kernel <<<B_*T_,            128, 0, stream>>>(x, W_Q, b_Q, W_K, b_K, xn, Qb, Kb);
    v_gemm_kernel<<<B_*H_*8,          256, 0, stream>>>(xn, W_V, b_V, Vt);
    stats_kernel <<<B_*H_*4,          256, 0, stream>>>(Qb, Kb, Mb, Lb);
    attn_v_kernel<<<B_*H_*8,          256, 0, stream>>>(Qb, Kb, Mb, Lb, Vt, HO);
    merger_kernel<<<(B_*T_*EH_)/256,  256, 0, stream>>>(HO, x, W1, b1, W2, b2, out);
}
